// RIMAEBackbone_3058016714858
// MI455X (gfx1250) — matrix operand order repaired and checked by reading, not compiled, not run on hardware
//
#include <hip/hip_runtime.h>
#include <math.h>

// ---------------- constants ----------------
#define BB 32
#define GG 512
#define DD 384
#define NH 6
#define HD 64
#define LL 12
#define KKEEP 205
#define HO 192
#define FF 1536
#define ROWS (BB*KKEEP)          // 6560 tokens
#define KPAD 224                  // keys padded to 7 chunks of 32

typedef __attribute__((ext_vector_type(16))) __bf16        v16bf;
typedef __attribute__((ext_vector_type(8)))  float         v8f;
typedef __attribute__((ext_vector_type(8)))  unsigned int  v8u;
typedef int v4i_ __attribute__((vector_size(16)));

union FragU { v8u u; v16bf b; };

static __device__ inline unsigned short f2bf(float f) {
  unsigned u = __builtin_bit_cast(unsigned, f);
  unsigned r = (u + 0x7FFFu + ((u >> 16) & 1u)) >> 16;
  return (unsigned short)r;
}
static __device__ inline unsigned pack2(float lo, float hi) {
  return (unsigned)f2bf(lo) | ((unsigned)f2bf(hi) << 16);
}
static __device__ inline v16bf pack_frag(uint4 a, uint4 c) {
  FragU f;
  f.u[0]=a.x; f.u[1]=a.y; f.u[2]=a.z; f.u[3]=a.w;
  f.u[4]=c.x; f.u[5]=c.y; f.u[6]=c.z; f.u[7]=c.w;
  return f.b;
}
static __device__ inline v8f vzero8() {
  v8f z;
  #pragma unroll
  for (int i = 0; i < 8; i++) z[i] = 0.0f;
  return z;
}
static __device__ inline float gelu_f(float x) {
  return 0.5f * x * (1.0f + erff(x * 0.70710678118654752f));
}

// -------- CDNA5 async global->LDS (ASYNCcnt) --------
static __device__ inline void async_cp16(const unsigned short* g, unsigned short* l) {
#if __has_builtin(__builtin_amdgcn_global_load_async_to_lds_b128)
  __builtin_amdgcn_global_load_async_to_lds_b128(
      (__attribute__((address_space(1))) v4i_*)g,
      (__attribute__((address_space(3))) v4i_*)l, 0, 0);
#else
  unsigned loff = (unsigned)(unsigned long long)
      ((__attribute__((address_space(3))) unsigned short*)l);
  asm volatile("global_load_async_to_lds_b128 %0, %1, off"
               :: "v"(loff), "v"(g) : "memory");
#endif
}
static __device__ inline void wait_async0() {
#if __has_builtin(__builtin_amdgcn_s_wait_asynccnt)
  __builtin_amdgcn_s_wait_asynccnt(0);
#else
  asm volatile("s_wait_asynccnt 0" ::: "memory");
#endif
}

// ---------------- fp32 -> bf16 weight conversion (transposed: W[K,N] -> Wt[N,K]) ----------------
__global__ __launch_bounds__(256) void f2bf_T_kernel(const float* __restrict__ w,
                                                     unsigned short* __restrict__ wt,
                                                     int K, int N) {
  size_t base = (size_t)blockIdx.y * K * N;
  int idx = blockIdx.x * 256 + threadIdx.x;
  if (idx >= K * N) return;
  int k = idx % K, n = idx / K;
  wt[base + (size_t)n * K + k] = f2bf(w[base + (size_t)k * N + n]);
}

// ---------------- gather visible tokens ----------------
__global__ __launch_bounds__(256) void gather_kernel(const float* __restrict__ x,
                                                     const int* __restrict__ ids,
                                                     float* __restrict__ xc) {
  int idx = blockIdx.x * 256 + threadIdx.x;
  const int total = ROWS * DD;
  if (idx >= total) return;
  int d = idx % DD;
  int t = idx / DD;
  int r = t % KKEEP;
  int b = t / KKEEP;
  xc[idx] = x[((size_t)b * GG + ids[b * KKEEP + r]) * DD + d];
}

// ---------------- orientation bias (fused MLP + mean + gather) ----------------
__global__ __launch_bounds__(256) void orient_kernel(const float* __restrict__ frames,
                                                     const int* __restrict__ ids,
                                                     const float* __restrict__ w1,
                                                     const float* __restrict__ b1,
                                                     const float* __restrict__ w2,
                                                     const float* __restrict__ b2,
                                                     const float* __restrict__ oscale,
                                                     float* __restrict__ vb) {
  __shared__ float s_w1[9 * HO];
  __shared__ float s_b1[HO];
  __shared__ float s_w2[HO * NH];
  __shared__ float s_b2[NH];
  __shared__ float s_Rp[9];
  __shared__ float s_row[NH][GG];
  __shared__ float s_red[256];
  __shared__ float s_mean[NH];

  int b = blockIdx.x / KKEEP;
  int j = blockIdx.x % KKEEP;
  int tid = threadIdx.x;

  for (int i = tid; i < 9 * HO; i += 256) s_w1[i] = w1[i];
  for (int i = tid; i < HO; i += 256) s_b1[i] = b1[i];
  for (int i = tid; i < HO * NH; i += 256) s_w2[i] = w2[i];
  if (tid < NH) s_b2[tid] = b2[tid];
  int p = ids[b * KKEEP + j];
  if (tid < 9) s_Rp[tid] = frames[((size_t)b * GG + p) * 9 + tid];
  __syncthreads();

  float Rp[9];
  #pragma unroll
  for (int i = 0; i < 9; i++) Rp[i] = s_Rp[i];

  for (int q = tid; q < GG; q += 256) {
    float Rq[9];
    #pragma unroll
    for (int i = 0; i < 9; i++) Rq[i] = frames[((size_t)b * GG + q) * 9 + i];
    float rel[9];
    #pragma unroll
    for (int jj = 0; jj < 3; jj++)
      #pragma unroll
      for (int kk = 0; kk < 3; kk++)
        rel[jj * 3 + kk] = Rp[0 + jj] * Rq[0 + kk] + Rp[3 + jj] * Rq[3 + kk] + Rp[6 + jj] * Rq[6 + kk];
    float acc[NH];
    #pragma unroll
    for (int hh = 0; hh < NH; hh++) acc[hh] = s_b2[hh];
    for (int u = 0; u < HO; u++) {
      float a = s_b1[u];
      #pragma unroll
      for (int r = 0; r < 9; r++) a += rel[r] * s_w1[r * HO + u];
      float g = gelu_f(a);
      #pragma unroll
      for (int hh = 0; hh < NH; hh++) acc[hh] += g * s_w2[u * NH + hh];
    }
    #pragma unroll
    for (int hh = 0; hh < NH; hh++) s_row[hh][q] = acc[hh];
  }
  __syncthreads();

  for (int hh = 0; hh < NH; hh++) {
    s_red[tid] = s_row[hh][tid] + s_row[hh][tid + 256];
    __syncthreads();
    for (int st = 128; st > 0; st >>= 1) {
      if (tid < st) s_red[tid] += s_red[tid + st];
      __syncthreads();
    }
    if (tid == 0) s_mean[hh] = s_red[0] * (1.0f / GG);
    __syncthreads();
  }
  float osc = oscale[0];
  for (int idx = tid; idx < NH * KKEEP; idx += 256) {
    int hh = idx / KKEEP, c = idx % KKEEP;
    int qg = ids[b * KKEEP + c];
    vb[(((size_t)b * NH + hh) * KKEEP + j) * KKEEP + c] = (s_row[hh][qg] - s_mean[hh]) * osc;
  }
}

// ---------------- layernorm (row per block), output bf16 or fp32 ----------------
__global__ __launch_bounds__(128) void ln_kernel(const float* __restrict__ x,
                                                 const float* __restrict__ g,
                                                 const float* __restrict__ bta,
                                                 unsigned short* __restrict__ outh,
                                                 float* __restrict__ outf) {
  int row = blockIdx.x;
  int tid = threadIdx.x;
  const float* xr = x + (size_t)row * DD;
  float v0 = xr[tid], v1 = xr[tid + 128], v2 = xr[tid + 256];
  __shared__ float red[128];
  __shared__ float s_mean;
  __shared__ float s_rstd;
  red[tid] = v0 + v1 + v2;
  __syncthreads();
  for (int st = 64; st > 0; st >>= 1) {
    if (tid < st) red[tid] += red[tid + st];
    __syncthreads();
  }
  if (tid == 0) s_mean = red[0] * (1.0f / DD);
  __syncthreads();
  float m = s_mean;
  float d0 = v0 - m, d1 = v1 - m, d2 = v2 - m;
  red[tid] = d0 * d0 + d1 * d1 + d2 * d2;
  __syncthreads();
  for (int st = 64; st > 0; st >>= 1) {
    if (tid < st) red[tid] += red[tid + st];
    __syncthreads();
  }
  if (tid == 0) s_rstd = rsqrtf(red[0] * (1.0f / DD) + 1e-5f);
  __syncthreads();
  float rs = s_rstd;
  float vv[3] = {d0, d1, d2};
  #pragma unroll
  for (int k = 0; k < 3; k++) {
    int d = tid + 128 * k;
    float y = vv[k] * rs * g[d] + bta[d];
    if (outh) outh[(size_t)row * DD + d] = f2bf(y);
    else      outf[(size_t)row * DD + d] = y;
  }
}

// ---------------- bf16 WMMA GEMM, async double-buffered LDS staging ----------------
// Wt is pre-transposed [N][K]. Block: 256 thr (8 waves), tile 128(M) x 64(N).
// mode 0: outh = bf16(acc + bias)           [QKV]
// mode 1: outf = resid + acc + bias (fp32)  [proj / mlp2; resid may alias outf]
// mode 2: outh = bf16(gelu(acc + bias))     [mlp1]
__global__ __launch_bounds__(256) void gemm_kernel(const unsigned short* __restrict__ A,
                                                   const unsigned short* __restrict__ Wt,
                                                   const float* __restrict__ bias,
                                                   const float* __restrict__ resid,
                                                   float* __restrict__ outf,
                                                   unsigned short* __restrict__ outh,
                                                   int M, int N, int K, int mode) {
  __shared__ unsigned short As[2][128][40];  // padded stride (80B) for bank spread
  __shared__ unsigned short Bs[2][64][40];
  int n0 = blockIdx.x * 64, m0 = blockIdx.y * 128;
  int tid = threadIdx.x;
  int wv = tid >> 5, lane = tid & 31, lh = lane >> 4, lm = lane & 15;
  int wm = wv >> 1, wn = wv & 1;   // 4 x 2 waves of 32x32

  v8f acc[2][2];
  #pragma unroll
  for (int mi = 0; mi < 2; mi++)
    #pragma unroll
    for (int ni = 0; ni < 2; ni++) acc[mi][ni] = vzero8();

  auto issue = [&](int buf, int kc) {
    // A tile 128x32: 512 16-byte chunks over 256 threads
    #pragma unroll
    for (int rr = 0; rr < 2; rr++) {
      int id = tid + rr * 256;
      int r = id >> 2, c8 = (id & 3) * 8;
      int gr = m0 + r; if (gr >= M) gr = M - 1;
      async_cp16(A + (size_t)gr * K + kc + c8, &As[buf][r][c8]);
    }
    // B tile 64x32: 256 chunks
    {
      int r = tid >> 2, c8 = (tid & 3) * 8;
      async_cp16(Wt + (size_t)(n0 + r) * K + kc + c8, &Bs[buf][r][c8]);
    }
  };

  int nk = K / 32;
  issue(0, 0);
  wait_async0();
  __syncthreads();

  for (int t = 0; t < nk; t++) {
    int buf = t & 1;
    if (t + 1 < nk) issue(buf ^ 1, (t + 1) * 32);

    v16bf afrag[2], bfrag[2];
    #pragma unroll
    for (int mi = 0; mi < 2; mi++) {
      int row = wm * 32 + mi * 16 + lm;
      uint4 c0 = *(const uint4*)&As[buf][row][8 * lh];
      uint4 c1 = *(const uint4*)&As[buf][row][16 + 8 * lh];
      afrag[mi] = pack_frag(c0, c1);
    }
    #pragma unroll
    for (int ni = 0; ni < 2; ni++) {
      int col = wn * 32 + ni * 16 + lm;
      uint4 d0 = *(const uint4*)&Bs[buf][col][8 * lh];
      uint4 d1 = *(const uint4*)&Bs[buf][col][16 + 8 * lh];
      bfrag[ni] = pack_frag(d0, d1);
    }
    #pragma unroll
    for (int mi = 0; mi < 2; mi++)
      #pragma unroll
      for (int ni = 0; ni < 2; ni++)
        acc[mi][ni] = __builtin_amdgcn_wmma_f32_16x16x32_bf16(
            false, afrag[mi], false, bfrag[ni], (short)0, acc[mi][ni], false, false);

    if (t + 1 < nk) wait_async0();
    __syncthreads();
  }

  #pragma unroll
  for (int mi = 0; mi < 2; mi++)
    #pragma unroll
    for (int ni = 0; ni < 2; ni++)
      #pragma unroll
      for (int j2 = 0; j2 < 8; j2++) {
        int row = m0 + wm * 32 + mi * 16 + j2 + 8 * lh;
        int col = n0 + wn * 32 + ni * 16 + lm;
        if (row < M) {
          float v = acc[mi][ni][j2] + bias[col];
          size_t o = (size_t)row * N + col;
          if (mode == 0)      outh[o] = f2bf(v);
          else if (mode == 1) outf[o] = resid[o] + v;
          else                outh[o] = f2bf(gelu_f(v));
        }
      }
}

// ---------------- fused attention per (qblock, h, b) ----------------
__global__ __launch_bounds__(128) void attn_kernel(const unsigned short* __restrict__ qkv,
                                                   const float* __restrict__ vb,
                                                   unsigned short* __restrict__ obuf) {
  extern __shared__ char smem[];
  unsigned short* Vt = (unsigned short*)smem;                 // [64][KPAD] bf16
  float* sc = (float*)(smem + (size_t)64 * KPAD * 2);         // 4 waves x [16][KPAD] f32

  int qblk = blockIdx.x, h = blockIdx.y, b = blockIdx.z;
  int tid = threadIdx.x;

  for (int i = tid; i < 64 * KPAD / 2; i += 128) ((unsigned*)Vt)[i] = 0u;
  __syncthreads();
  for (int i = tid; i < KKEEP * 64; i += 128) {
    int key = i >> 6, hd = i & 63;
    Vt[hd * KPAD + key] = qkv[(size_t)(b * KKEEP + key) * (3 * DD) + 2 * DD + h * HD + hd];
  }
  __syncthreads();

  int wv = tid >> 5, lane = tid & 31, lh = lane >> 4, lm = lane & 15;
  float* srow_base = sc + (size_t)wv * 16 * KPAD;
  int q0 = qblk * 64 + wv * 16;
  int qr = q0 + lm;
  int qc = qr < KKEEP ? qr : KKEEP - 1;

  const unsigned short* qp = qkv + (size_t)(b * KKEEP + qc) * (3 * DD) + h * HD;
  v16bf aq[2];
  #pragma unroll
  for (int s = 0; s < 2; s++) {
    uint4 c0 = *(const uint4*)(qp + s * 32 + 8 * lh);
    uint4 c1 = *(const uint4*)(qp + s * 32 + 16 + 8 * lh);
    aq[s] = pack_frag(c0, c1);
  }

  for (int kt = 0; kt < 13; kt++) {
    int key = kt * 16 + lm;
    int kc2 = key < KKEEP ? key : KKEEP - 1;
    const unsigned short* kp = qkv + (size_t)(b * KKEEP + kc2) * (3 * DD) + DD + h * HD;
    v8f acc = vzero8();
    #pragma unroll
    for (int s = 0; s < 2; s++) {
      uint4 d0 = *(const uint4*)(kp + s * 32 + 8 * lh);
      uint4 d1 = *(const uint4*)(kp + s * 32 + 16 + 8 * lh);
      v16bf bf = pack_frag(d0, d1);
      acc = __builtin_amdgcn_wmma_f32_16x16x32_bf16(false, aq[s], false, bf, (short)0, acc, false, false);
    }
    #pragma unroll
    for (int j2 = 0; j2 < 8; j2++)
      srow_base[(j2 + 8 * lh) * KPAD + kt * 16 + lm] = acc[j2];
  }
  __syncthreads();

  if (lane < 16) {
    int row = lane, q = q0 + row;
    float* sr = srow_base + row * KPAD;
    if (q < KKEEP) {
      const float* vr = vb + (((size_t)b * NH + h) * KKEEP + q) * KKEEP;
      float mx = -1e30f;
      for (int k = 0; k < KKEEP; k++) {
        float s = sr[k] * 0.125f + vr[k];
        sr[k] = s;
        mx = fmaxf(mx, s);
      }
      float sum = 0.0f;
      for (int k = 0; k < KKEEP; k++) { float e = expf(sr[k] - mx); sr[k] = e; sum += e; }
      float inv = 1.0f / sum;
      for (int k = 0; k < KKEEP; k++) sr[k] *= inv;
      for (int k = KKEEP; k < KPAD; k++) sr[k] = 0.0f;
    } else {
      for (int k = 0; k < KPAD; k++) sr[k] = 0.0f;
    }
  }
  __syncthreads();

  v8f oacc[4];
  #pragma unroll
  for (int ni = 0; ni < 4; ni++) oacc[ni] = vzero8();
  const float* prow = srow_base + lm * KPAD;
  for (int c = 0; c < 7; c++) {
    const float4* f0 = (const float4*)(prow + c * 32 + 8 * lh);
    const float4* f1 = (const float4*)(prow + c * 32 + 16 + 8 * lh);
    float4 a0 = f0[0], a1 = f0[1], b0 = f1[0], b1 = f1[1];
    FragU pf;
    pf.u[0] = pack2(a0.x, a0.y); pf.u[1] = pack2(a0.z, a0.w);
    pf.u[2] = pack2(a1.x, a1.y); pf.u[3] = pack2(a1.z, a1.w);
    pf.u[4] = pack2(b0.x, b0.y); pf.u[5] = pack2(b0.z, b0.w);
    pf.u[6] = pack2(b1.x, b1.y); pf.u[7] = pack2(b1.z, b1.w);
    v16bf pa = pf.b;
    #pragma unroll
    for (int ni = 0; ni < 4; ni++) {
      const unsigned short* vp = Vt + (size_t)(ni * 16 + lm) * KPAD + c * 32;
      v16bf vbf = pack_frag(*(const uint4*)(vp + 8 * lh), *(const uint4*)(vp + 16 + 8 * lh));
      oacc[ni] = __builtin_amdgcn_wmma_f32_16x16x32_bf16(false, pa, false, vbf, (short)0, oacc[ni], false, false);
    }
  }
  #pragma unroll
  for (int ni = 0; ni < 4; ni++)
    #pragma unroll
    for (int j2 = 0; j2 < 8; j2++) {
      int q = q0 + j2 + 8 * lh;
      if (q < KKEEP)
        obuf[(size_t)(b * KKEEP + q) * DD + h * HD + ni * 16 + lm] = f2bf(oacc[ni][j2]);
    }
}

// ---------------- host orchestration ----------------
extern "C" void kernel_launch(void* const* d_in, const int* in_sizes, int n_in,
                              void* d_out, int out_size, void* d_ws, size_t ws_size,
                              hipStream_t stream) {
  (void)in_sizes; (void)n_in; (void)out_size; (void)ws_size;
  const float* x        = (const float*)d_in[0];
  const float* frames   = (const float*)d_in[1];
  const int*   ids      = (const int*)d_in[2];
  const float* om_w1    = (const float*)d_in[3];
  const float* om_b1    = (const float*)d_in[4];
  const float* om_w2    = (const float*)d_in[5];
  const float* om_b2    = (const float*)d_in[6];
  const float* oscale   = (const float*)d_in[7];
  const float* ln1_g    = (const float*)d_in[8];
  const float* ln1_b    = (const float*)d_in[9];
  const float* w_qkv    = (const float*)d_in[10];
  const float* b_qkv    = (const float*)d_in[11];
  const float* w_proj   = (const float*)d_in[12];
  const float* b_proj   = (const float*)d_in[13];
  const float* ln2_g    = (const float*)d_in[14];
  const float* ln2_b    = (const float*)d_in[15];
  const float* w_m1     = (const float*)d_in[16];
  const float* b_m1     = (const float*)d_in[17];
  const float* w_m2     = (const float*)d_in[18];
  const float* b_m2     = (const float*)d_in[19];
  const float* lnf_g    = (const float*)d_in[20];
  const float* lnf_b    = (const float*)d_in[21];
  float* out = (float*)d_out;

  char* ws = (char*)d_ws;
  size_t off = 0;
  auto nxt = [&](size_t bytes) { size_t o = off; off += (bytes + 255) & ~(size_t)255; return o; };

  unsigned short* wqb  = (unsigned short*)(ws + nxt((size_t)LL * DD * 3 * DD * 2));
  unsigned short* wpb  = (unsigned short*)(ws + nxt((size_t)LL * DD * DD * 2));
  unsigned short* w1b  = (unsigned short*)(ws + nxt((size_t)LL * DD * FF * 2));
  unsigned short* w2b  = (unsigned short*)(ws + nxt((size_t)LL * FF * DD * 2));
  float*          xc   = (float*)(ws + nxt((size_t)ROWS * DD * 4));
  unsigned short* hn   = (unsigned short*)(ws + nxt((size_t)ROWS * DD * 2));
  unsigned short* qkvb = (unsigned short*)(ws + nxt((size_t)ROWS * 3 * DD * 2));
  unsigned short* ob   = (unsigned short*)(ws + nxt((size_t)ROWS * DD * 2));
  unsigned short* hid  = (unsigned short*)(ws + nxt((size_t)ROWS * FF * 2));
  float*          vbw  = (float*)(ws + nxt((size_t)BB * NH * KKEEP * KKEEP * 4));

  // 1) convert weights to bf16, transposed to [N][K] for async row staging
  f2bf_T_kernel<<<dim3((DD * 3 * DD + 255) / 256, LL), 256, 0, stream>>>(w_qkv, wqb, DD, 3 * DD);
  f2bf_T_kernel<<<dim3((DD * DD + 255) / 256, LL), 256, 0, stream>>>(w_proj, wpb, DD, DD);
  f2bf_T_kernel<<<dim3((DD * FF + 255) / 256, LL), 256, 0, stream>>>(w_m1, w1b, DD, FF);
  f2bf_T_kernel<<<dim3((FF * DD + 255) / 256, LL), 256, 0, stream>>>(w_m2, w2b, FF, DD);

  // 2) gather visible tokens; 3) orientation bias
  gather_kernel<<<(ROWS * DD + 255) / 256, 256, 0, stream>>>(x, ids, xc);
  orient_kernel<<<BB * KKEEP, 256, 0, stream>>>(frames, ids, om_w1, om_b1, om_w2, om_b2, oscale, vbw);

  size_t attn_lds = (size_t)64 * KPAD * 2 + (size_t)4 * 16 * KPAD * 4;
  int mgrid = (ROWS + 127) / 128;

  for (int l = 0; l < LL; l++) {
    ln_kernel<<<ROWS, 128, 0, stream>>>(xc, ln1_g + l * DD, ln1_b + l * DD, hn, nullptr);
    gemm_kernel<<<dim3(3 * DD / 64, mgrid), 256, 0, stream>>>(
        hn, wqb + (size_t)l * DD * 3 * DD, b_qkv + l * 3 * DD,
        nullptr, nullptr, qkvb, ROWS, 3 * DD, DD, 0);
    attn_kernel<<<dim3(4, NH, BB), 128, attn_lds, stream>>>(qkvb, vbw, ob);
    gemm_kernel<<<dim3(DD / 64, mgrid), 256, 0, stream>>>(
        ob, wpb + (size_t)l * DD * DD, b_proj + l * DD,
        xc, xc, nullptr, ROWS, DD, DD, 1);
    ln_kernel<<<ROWS, 128, 0, stream>>>(xc, ln2_g + l * DD, ln2_b + l * DD, hn, nullptr);
    gemm_kernel<<<dim3(FF / 64, mgrid), 256, 0, stream>>>(
        hn, w1b + (size_t)l * DD * FF, b_m1 + l * FF,
        nullptr, nullptr, hid, ROWS, FF, DD, 2);
    gemm_kernel<<<dim3(DD / 64, mgrid), 256, 0, stream>>>(
        hid, w2b + (size_t)l * FF * DD, b_m2 + l * DD,
        xc, xc, nullptr, ROWS, DD, FF, 1);
  }

  ln_kernel<<<ROWS, 128, 0, stream>>>(xc, lnf_g, lnf_b, nullptr, out);
}
